// AttentionReadout_24910810316948
// MI455X (gfx1250) — matrix operand order repaired and checked by reading, not compiled, not run on hardware
//
#include <hip/hip_runtime.h>
#include <hip/hip_bf16.h>

typedef __attribute__((ext_vector_type(16))) _Float16 v16h;
typedef __attribute__((ext_vector_type(8)))  _Float16 v8h;
typedef __attribute__((ext_vector_type(8)))  float    v8f;

// Fixed problem geometry (matches reference harness).
constexpr int NB    = 64;    // graphs
constexpr int ND    = 128;   // node feature dim
constexpr int HID   = 256;   // per-head hidden
constexpr int HEADS = 8;
constexpr int NPAD  = 96;    // dense padded length
constexpr int DTOT  = HID * HEADS; // 2048

// ---------------- WMMA fragment helpers (ISA 7.12.2 layouts) ----------------
// A (16x32 f16): lane l (<16) holds row M=l, halves j=0..7 -> K=j, j=8..15 -> K=16+(j-8);
// lane l (>=16) same rows, K offset by 8 / 24. Two contiguous 8-half (16B) loads.
static __device__ __forceinline__ v16h ld_afrag(const _Float16* __restrict__ base,
                                                int stride, int m0, int k0, int lane) {
  const int row = m0 + (lane & 15);
  const int off = (lane >> 4) << 3;            // 0 or 8
  const _Float16* p = base + (size_t)row * stride + k0 + off;
  v8h lo = *(const v8h*)(p);
  v8h hi = *(const v8h*)(p + 16);
  v16h a;
#pragma unroll
  for (int i = 0; i < 8; ++i) { a[i] = lo[i]; a[i + 8] = hi[i]; }
  return a;
}

// B (32x16 f16): lane l holds column n=l&15; same K order as A (j<8 -> K=j+8h, j>=8 -> K=16+8h+(j-8)).
// Source buffer is N-major ("Bt[n][k]"), so both 8-half groups are contiguous.
static __device__ __forceinline__ v16h ld_bfrag(const _Float16* __restrict__ base,
                                                int stride, int n0, int k0, int lane) {
  const int row = n0 + (lane & 15);
  const int off = (lane >> 4) << 3;            // 0 or 8
  const _Float16* p = base + (size_t)row * stride + k0 + off;
  v8h lo = *(const v8h*)(p);
  v8h hi = *(const v8h*)(p + 16);
  v16h b;
#pragma unroll
  for (int i = 0; i < 8; ++i) { b[i] = lo[i]; b[i + 8] = hi[i]; }
  return b;
}

static __device__ __forceinline__ v8f wmma16(v16h a, v16h b, v8f c) {
  return __builtin_amdgcn_wmma_f32_16x16x32_f16(false, a, false, b, (short)0, c,
                                                false, false);
}

// out[m][c] = sum_k Xs[m][k] * W[k][c] + bias[c], W given N-major f16 [HID][ND].
static __device__ __forceinline__ void proj_gemm(const _Float16* __restrict__ Xs,
                                                 const _Float16* __restrict__ WtHead,
                                                 const float* __restrict__ biasHead,
                                                 _Float16* __restrict__ dst,
                                                 int wave, int lane) {
  for (int t = wave; t < 6 * 16; t += 8) {
    const int m0 = (t % 6) * 16, n0 = (t / 6) * 16;
    v8f acc = {};
#pragma unroll
    for (int kk = 0; kk < ND; kk += 32)
      acc = wmma16(ld_afrag(Xs, ND, m0, kk, lane),
                   ld_bfrag(WtHead, ND, n0, kk, lane), acc);
    const int col  = n0 + (lane & 15);
    const int rb   = m0 + ((lane >> 4) << 3);
    const float bs = biasHead[col];
#pragma unroll
    for (int r = 0; r < 8; ++r)
      dst[(size_t)(rb + r) * HID + col] = (_Float16)(acc[r] + bs);
  }
}

// Same projection but store transposed Vt[c][m] (N-major for the P@V GEMM);
// each lane's 8 rows are contiguous -> one packed 16B LDS store per tile.
static __device__ __forceinline__ void proj_gemm_vT(const _Float16* __restrict__ Xs,
                                                    const _Float16* __restrict__ WtHead,
                                                    const float* __restrict__ biasHead,
                                                    _Float16* __restrict__ Vt,
                                                    int wave, int lane) {
  for (int t = wave; t < 6 * 16; t += 8) {
    const int m0 = (t % 6) * 16, n0 = (t / 6) * 16;
    v8f acc = {};
#pragma unroll
    for (int kk = 0; kk < ND; kk += 32)
      acc = wmma16(ld_afrag(Xs, ND, m0, kk, lane),
                   ld_bfrag(WtHead, ND, n0, kk, lane), acc);
    const int col  = n0 + (lane & 15);
    const int rb   = m0 + ((lane >> 4) << 3);
    const float bs = biasHead[col];
    v8h pk;
#pragma unroll
    for (int r = 0; r < 8; ++r) pk[r] = (_Float16)(acc[r] + bs);
    *(v8h*)(Vt + (size_t)col * NPAD + rb) = pk;
  }
}

// ---------------- Kernels ----------------
extern "C" __global__ void ar_starts_kernel(const int* __restrict__ batch, int total,
                                            int* __restrict__ starts) {
  const int b = threadIdx.x;
  if (b <= NB) {            // starts[NB] = total (batch values are all < NB)
    int lo = 0, hi = total;
    while (lo < hi) { int mid = (lo + hi) >> 1; if (batch[mid] < b) lo = mid + 1; else hi = mid; }
    starts[b] = lo;
  }
}

// W f32 [ND][DTOT] -> Wt f16 [DTOT][ND] (N-major for b128 fragment loads).
extern "C" __global__ void ar_cvtw_kernel(const float* __restrict__ W,
                                          _Float16* __restrict__ Wt) {
  const int i = blockIdx.x * blockDim.x + threadIdx.x;
  if (i < DTOT * ND) {
    const int col = i / ND, row = i % ND;
    Wt[i] = (_Float16)W[(size_t)row * DTOT + col];
  }
}

extern "C" __global__ void ar_zero_kernel(float* __restrict__ p, int n) {
  const int i = blockIdx.x * blockDim.x + threadIdx.x;
  if (i < n) p[i] = 0.0f;
}

// One workgroup per (head, graph): fully LDS-resident fused attention.
extern "C" __global__ __launch_bounds__(256)
void ar_attn_kernel(const float* __restrict__ x, const int* __restrict__ starts,
                    const _Float16* __restrict__ Wqt, const _Float16* __restrict__ Wkt,
                    const _Float16* __restrict__ Wvt,
                    const float* __restrict__ bq, const float* __restrict__ bk,
                    const float* __restrict__ bv, float* __restrict__ hsum) {
  const int h = blockIdx.x, b = blockIdx.y;
  extern __shared__ char smem[];
  _Float16* Xs = (_Float16*)(smem);                           // [96][128] f16  24576B
  _Float16* Qs = (_Float16*)(smem + 24576);                   // [96][256] f16  49152B
  _Float16* Ks = (_Float16*)(smem + 24576 + 49152);           // [96][256] f16  49152B
  float*    Ss = (float*)   (smem + 24576 + 98304);           // [96][96]  f32  36864B
  _Float16* Ps = (_Float16*)(smem + 24576 + 98304 + 36864);   // [96][96]  f16  18432B
  _Float16* Vt = Qs;                                          // [256][96] overlays Qs

  const int tid = threadIdx.x, lane = tid & 31, wave = tid >> 5;
  const int start = starts[b];
  const int count = starts[b + 1] - start;

  // Dense-batch tile: valid rows from x, padded rows zero (so k=bk, v=bv there).
  for (int i = tid; i < NPAD * ND; i += 256) {
    const int r = i >> 7, c = i & (ND - 1);
    Xs[i] = (_Float16)((r < count) ? x[(size_t)(start + r) * ND + c] : 0.0f);
  }
  __syncthreads();

  proj_gemm(Xs, Wqt + (size_t)h * HID * ND, bq + h * HID, Qs, wave, lane);
  proj_gemm(Xs, Wkt + (size_t)h * HID * ND, bk + h * HID, Ks, wave, lane);
  __syncthreads();

  // S = (Q @ K^T) * 1/sqrt(NODE_DIM)
  const float scale = 0.08838834764831845f;
  for (int t = wave; t < 36; t += 8) {
    const int m0 = (t % 6) * 16, n0 = (t / 6) * 16;
    v8f acc = {};
#pragma unroll
    for (int kk = 0; kk < HID; kk += 32)
      acc = wmma16(ld_afrag(Qs, HID, m0, kk, lane),
                   ld_bfrag(Ks, HID, n0, kk, lane), acc);
    const int col = n0 + (lane & 15);
    const int rb  = m0 + ((lane >> 4) << 3);
#pragma unroll
    for (int r = 0; r < 8; ++r) Ss[(rb + r) * NPAD + col] = acc[r] * scale;
  }
  __syncthreads();

  // Softmax over keys; reference masks the QUERY axis -> padded rows uniform 1/96.
  if (tid < NPAD) {
    float* row = Ss + tid * NPAD;
    _Float16* prow = Ps + tid * NPAD;
    if (tid < count) {
      float mx = -1e30f;
      for (int m = 0; m < NPAD; ++m) mx = fmaxf(mx, row[m]);
      float s = 0.0f;
      for (int m = 0; m < NPAD; ++m) s += __expf(row[m] - mx);
      const float inv = 1.0f / s;
      for (int m = 0; m < NPAD; ++m) prow[m] = (_Float16)(__expf(row[m] - mx) * inv);
    } else {
      const _Float16 u = (_Float16)(1.0f / NPAD);
      for (int m = 0; m < NPAD; ++m) prow[m] = u;
    }
  }
  __syncthreads();

  // V, stored transposed into the (now free) Q region.
  proj_gemm_vT(Xs, Wvt + (size_t)h * HID * ND, bv + h * HID, Vt, wave, lane);
  __syncthreads();

  // O = P @ V; reduce over all 96 query rows and accumulate per-column sums.
  float* hs = hsum + (size_t)b * DTOT + h * HID;
  for (int t = wave; t < 6 * 16; t += 8) {
    const int m0 = (t % 6) * 16, n0 = (t / 6) * 16;
    v8f acc = {};
#pragma unroll
    for (int kk = 0; kk < NPAD; kk += 32)
      acc = wmma16(ld_afrag(Ps, NPAD, m0, kk, lane),
                   ld_bfrag(Vt, NPAD, n0, kk, lane), acc);
    float partial = 0.0f;
#pragma unroll
    for (int r = 0; r < 8; ++r) partial += acc[r];
    atomicAdd(&hs[n0 + (lane & 15)], partial);     // lanes l and l+16 share a column
  }
}

// readout[b] = hsum[b] @ Wo + NPAD * bo
extern "C" __global__ void ar_readout_kernel(const float* __restrict__ hsum,
                                             const float* __restrict__ Wo,
                                             const float* __restrict__ bo,
                                             float* __restrict__ out) {
  const int b = blockIdx.x, c = threadIdx.x;   // 128 threads
  float acc = (float)NPAD * bo[c];
  const float* hrow = hsum + (size_t)b * DTOT;
  for (int k = 0; k < DTOT; ++k) acc += hrow[k] * Wo[(size_t)k * ND + c];
  out[b * ND + c] = acc;
}

// ---------------- Host launch ----------------
extern "C" void kernel_launch(void* const* d_in, const int* in_sizes, int n_in,
                              void* d_out, int out_size, void* d_ws, size_t ws_size,
                              hipStream_t stream) {
  const float* x     = (const float*)d_in[0];
  const int*   batch = (const int*)d_in[1];
  const float* Wq = (const float*)d_in[2]; const float* bq = (const float*)d_in[3];
  const float* Wk = (const float*)d_in[4]; const float* bk = (const float*)d_in[5];
  const float* Wv = (const float*)d_in[6]; const float* bv = (const float*)d_in[7];
  const float* Wo = (const float*)d_in[8]; const float* bo = (const float*)d_in[9];
  float* out = (float*)d_out;
  const int total = in_sizes[1];

  // Workspace layout (all 512B-aligned):
  //   [0,512)                       : starts[65]
  //   +512  + 0*524288             : Wq^T f16 [2048][128]
  //   +512  + 1*524288             : Wk^T f16
  //   +512  + 2*524288             : Wv^T f16
  //   +512  + 3*524288             : hsum f32 [64][2048]
  char* ws = (char*)d_ws;
  int*      starts = (int*)ws;
  _Float16* Wqt = (_Float16*)(ws + 512);
  _Float16* Wkt = (_Float16*)(ws + 512 + 1 * 524288);
  _Float16* Wvt = (_Float16*)(ws + 512 + 2 * 524288);
  float*    hsum = (float*)  (ws + 512 + 3 * 524288);

  ar_starts_kernel<<<1, 128, 0, stream>>>(batch, total, starts);
  const int cvtN = DTOT * ND;
  ar_cvtw_kernel<<<(cvtN + 255) / 256, 256, 0, stream>>>(Wq, Wqt);
  ar_cvtw_kernel<<<(cvtN + 255) / 256, 256, 0, stream>>>(Wk, Wkt);
  ar_cvtw_kernel<<<(cvtN + 255) / 256, 256, 0, stream>>>(Wv, Wvt);
  ar_zero_kernel<<<(NB * DTOT + 255) / 256, 256, 0, stream>>>(hsum, NB * DTOT);

  const int SMEM = 24576 + 49152 + 49152 + 36864 + 18432;   // 178176 B < 320 KB LDS
  hipFuncSetAttribute(reinterpret_cast<const void*>(ar_attn_kernel),
                      hipFuncAttributeMaxDynamicSharedMemorySize, SMEM);
  ar_attn_kernel<<<dim3(HEADS, NB), 256, SMEM, stream>>>(x, starts, Wqt, Wkt, Wvt,
                                                         bq, bk, bv, hsum);

  ar_readout_kernel<<<NB, ND, 0, stream>>>(hsum, Wo, bo, out);
}
